// S4DKernel_9801115369843
// MI455X (gfx1250) — matrix operand order repaired and checked by reading, not compiled, not run on hardware
//
#include <hip/hip_runtime.h>
#include <hip/hip_bf16.h>

// S4D kernel materialization for gfx1250 (MI455X, wave32, WMMA).
// K[h, l1*64+l0] = P_h (64x64) @ Q_h (64x64), built per-block, WMMA f16->f32.

typedef __attribute__((ext_vector_type(16))) _Float16 v16h;
typedef __attribute__((ext_vector_type(8)))  _Float16 v8h;
typedef __attribute__((ext_vector_type(2)))  _Float16 v2h;
typedef __attribute__((ext_vector_type(8)))  float    v8f;

#define H_DIM  1024
#define NHALF  32      // N/2 complex modes
#define L_LEN  4096
#define KDIM   64      // contraction dim = 2*NHALF
#define PSTR   72      // padded LDS row stride in halfs (multiple of 8)

__global__ __launch_bounds__(128) void s4d_vandermonde_wmma(
    const float* __restrict__ log_dt,
    const float* __restrict__ C_real,     // (1,H,32,2)
    const float* __restrict__ log_A_real, // (H,32)
    const float* __restrict__ A_imag,     // (H,32)
    float* __restrict__ out)              // (1,H,4096)
{
    __shared__ __align__(16) _Float16 Pl[64 * PSTR]; // P[l1][k]  row-major in k
    __shared__ __align__(16) _Float16 Qt[64 * PSTR]; // Q^T: Qt[l0][k]

    const int h    = blockIdx.x;
    const int tid  = threadIdx.x;
    const int n    = tid & 31;   // mode index
    const int c    = tid >> 5;   // chunk / wave id (0..3)

    // ---------------- Phase 1: build P and Qt in LDS ----------------
    {
        const float dt   = __expf(log_dt[h]);
        const float Ar   = -__expf(log_A_real[h * NHALF + n]);
        const float Ai   = A_imag[h * NHALF + n];
        const float dtAr = Ar * dt, dtAi = Ai * dt;
        // denom = 1 - dtA/2
        const float dr  = 1.0f - 0.5f * dtAr;
        const float di  = -0.5f * dtAi;
        const float inv = 1.0f / (dr * dr + di * di);
        // B = dt / denom
        const float Br =  dt * dr * inv;
        const float Bi = -dt * di * inv;
        const float Cr = C_real[h * (2 * NHALF) + 2 * n + 0];
        const float Ci = C_real[h * (2 * NHALF) + 2 * n + 1];
        const float dCr = Cr * Br - Ci * Bi;
        const float dCi = Cr * Bi + Ci * Br;
        // dA = (1 + dtA/2) / (1 - dtA/2)
        const float nr  = 1.0f + 0.5f * dtAr;
        const float ni  = 0.5f * dtAi;
        const float dAr = (nr * dr + ni * di) * inv;
        const float dAi = (ni * dr - nr * di) * inv;

        // s16 = dA^16 via 4 squarings
        float sr = dAr, si = dAi;
        #pragma unroll
        for (int i = 0; i < 4; ++i) { float t = sr*sr - si*si; si = 2.0f*sr*si; sr = t; }

        // Q chunk start: z = dA^(16*c)
        float zr = 1.0f, zi = 0.0f;
        for (int i = 0; i < c; ++i) { float t = zr*sr - zi*si; zi = zr*si + zi*sr; zr = t; }
        // Qt rows l0 = 16c .. 16c+15 ; this thread fills columns (2n, 2n+1)
        float qr = zr, qi = zi;
        #pragma unroll
        for (int l = 0; l < 16; ++l) {
            const int row = c * 16 + l;
            v2h q; q[0] = (_Float16)qr; q[1] = (_Float16)qi;
            *(v2h*)(Qt + row * PSTR + 2 * n) = q;                 // ds_store_b32
            float t = qr*dAr - qi*dAi; qi = qr*dAi + qi*dAr; qr = t;
        }

        // W = dA^64 (2 more squarings), G = W^16 = dA^1024 (4 squarings)
        float wr = sr, wi = si;
        #pragma unroll
        for (int i = 0; i < 2; ++i) { float t = wr*wr - wi*wi; wi = 2.0f*wr*wi; wr = t; }
        float gr = wr, gi = wi;
        #pragma unroll
        for (int i = 0; i < 4; ++i) { float t = gr*gr - gi*gi; gi = 2.0f*gr*gi; gr = t; }
        // P chunk start: u = dC * G^c
        float ur = dCr, ui = dCi;
        for (int i = 0; i < c; ++i) { float t = ur*gr - ui*gi; ui = ur*gi + ui*gr; ur = t; }
        // P rows l1 = 16c .. 16c+15, columns (2n, 2n+1) = (2Re u, -2Im u)
        #pragma unroll
        for (int l = 0; l < 16; ++l) {
            const int row = c * 16 + l;
            v2h p; p[0] = (_Float16)(2.0f * ur); p[1] = (_Float16)(-2.0f * ui);
            *(v2h*)(Pl + row * PSTR + 2 * n) = p;                 // ds_store_b32
            float t = ur*wr - ui*wi; ui = ur*wi + ui*wr; ur = t;
        }
    }
    __syncthreads();

    // ---------------- Phase 2: 64x64x64 GEMM via WMMA ----------------
    // Wave `c` computes D rows l1 in [16c, 16c+16), looping 4 tiles over l0.
    const int lane = tid & 31;
    const int lmod = lane & 15;
    const int lhi  = lane >> 4;   // K sub-group select per ISA VGPR layout

    // A fragments (16x32 f16): lane holds row M=lmod; element e<8 -> K=kb+8*lhi+e,
    // e>=8 -> K=kb+16+8*lhi+(e-8). Contiguous 8-half runs => ds_load_b128.
    const int arow = c * 16 + lmod;
    v16h a0, a1;
    {
        const _Float16* ap = Pl + arow * PSTR;
        v8h x0 = *(const v8h*)(ap +      lhi * 8);
        v8h x1 = *(const v8h*)(ap + 16 + lhi * 8);
        v8h x2 = *(const v8h*)(ap + 32 + lhi * 8);
        v8h x3 = *(const v8h*)(ap + 48 + lhi * 8);
        #pragma unroll
        for (int i = 0; i < 8; ++i) {
            a0[i] = x0[i]; a0[i + 8] = x1[i];
            a1[i] = x2[i]; a1[i + 8] = x3[i];
        }
    }

    #pragma unroll
    for (int tn = 0; tn < 4; ++tn) {
        // B fragments (32x16 f16): lane holds column N=lmod; element e<8 -> K=kb+8*lhi+e, e>=8 -> K=kb+16+8*lhi+(e-8).
        const int brow = tn * 16 + lmod;       // l0 column, Qt is [l0][K]
        const _Float16* bp = Qt + brow * PSTR;
        v16h b0, b1;
        {
            v8h y0 = *(const v8h*)(bp +      lhi * 8);
            v8h y1 = *(const v8h*)(bp + 16 + lhi * 8);
            v8h y2 = *(const v8h*)(bp + 32 + lhi * 8);
            v8h y3 = *(const v8h*)(bp + 48 + lhi * 8);
            #pragma unroll
            for (int i = 0; i < 8; ++i) {
                b0[i] = y0[i]; b0[i + 8] = y1[i];
                b1[i] = y2[i]; b1[i + 8] = y3[i];
            }
        }
        v8f acc = {};
        acc = __builtin_amdgcn_wmma_f32_16x16x32_f16(false, a0, false, b0,
                                                     (short)0, acc, false, false);
        acc = __builtin_amdgcn_wmma_f32_16x16x32_f16(false, a1, false, b1,
                                                     (short)0, acc, false, false);
        // D layout: VGPR r -> (l1 = 16c + 8*lhi + r, l0 = 16*tn + lmod)
        float* op = out + (size_t)h * L_LEN + tn * 16 + lmod;
        const int l1base = c * 16 + lhi * 8;
        #pragma unroll
        for (int r = 0; r < 8; ++r)
            op[(l1base + r) * 64] = acc[r];
    }
}

extern "C" void kernel_launch(void* const* d_in, const int* in_sizes, int n_in,
                              void* d_out, int out_size, void* d_ws, size_t ws_size,
                              hipStream_t stream) {
    (void)in_sizes; (void)n_in; (void)d_ws; (void)ws_size; (void)out_size;
    const float* log_dt     = (const float*)d_in[0];
    const float* C_real     = (const float*)d_in[1];
    const float* log_A_real = (const float*)d_in[2];
    const float* A_imag     = (const float*)d_in[3];
    float* out = (float*)d_out;
    s4d_vandermonde_wmma<<<dim3(H_DIM), dim3(128), 0, stream>>>(
        log_dt, C_real, log_A_real, A_imag, out);
}
